// LigerLMHeadJSD_41180146434370
// MI455X (gfx1250) — matrix operand order repaired and checked by reading, not compiled, not run on hardware
//
#include <hip/hip_runtime.h>

#define BT 4096
#define H  2048
#define V  32000
#define BM 64
#define BN 64
#define BK 64
#define BKP 72            // padded LDS row stride (elements): 144B rows
#define NSPLIT 20
#define VC (V / NSPLIT)   // 1600 columns per chunk
#define NTILES (VC / BN)  // 25 n-tiles per chunk
#define NBLK (BT / BM)    // 64 row blocks
#define KITERS (H / BK)   // 32

typedef __attribute__((ext_vector_type(16))) __bf16 v16bf;
typedef __attribute__((ext_vector_type(8)))  float  v8f;

#if __has_builtin(__builtin_amdgcn_global_load_async_to_lds_b128)
#define HAVE_ASYNC 1
typedef int av4i __attribute__((vector_size(16)));  // matches builtin param pointee
#else
#define HAVE_ASYNC 0
#endif

union Frag {
  v16bf v;
  float4 q[2];
};

__device__ __forceinline__ unsigned pk_bf16(float a, float b) {
  unsigned ua = __float_as_uint(a), ub = __float_as_uint(b);
  ua += 0x7fffu + ((ua >> 16) & 1u);   // round-to-nearest-even fp32 -> bf16
  ub += 0x7fffu + ((ub >> 16) & 1u);
  return (ua >> 16) | (ub & 0xffff0000u);
}

// One-time fp32 -> bf16 conversion sweep: each thread converts 16 elements.
__global__ __launch_bounds__(256) void cvt_bf16_kernel(const float* __restrict__ src,
                                                       unsigned short* __restrict__ dst) {
  const size_t i = ((size_t)blockIdx.x * 256 + threadIdx.x) * 16;
  float4 f0 = *(const float4*)(src + i + 0);
  float4 f1 = *(const float4*)(src + i + 4);
  float4 f2 = *(const float4*)(src + i + 8);
  float4 f3 = *(const float4*)(src + i + 12);
  uint4 p0 = { pk_bf16(f0.x, f0.y), pk_bf16(f0.z, f0.w),
               pk_bf16(f1.x, f1.y), pk_bf16(f1.z, f1.w) };
  uint4 p1 = { pk_bf16(f2.x, f2.y), pk_bf16(f2.z, f2.w),
               pk_bf16(f3.x, f3.y), pk_bf16(f3.z, f3.w) };
  *(uint4*)(dst + i)     = p0;
  *(uint4*)(dst + i + 8) = p1;
}

// ---- Tile staging: 64 rows x 64 bf16 cols; 256 threads = 64 rows x 4 segs of 16.
#if HAVE_ASYNC
__device__ __forceinline__ void load_tile_async(unsigned short* __restrict__ lds,
                                                const unsigned short* __restrict__ g,
                                                int t) {
  const int row = t >> 2;
  const int seg = (t & 3) * 16;
  const unsigned short* src = g + (size_t)row * H + seg;
  unsigned short* dst = lds + (size_t)row * BKP + seg;
  __builtin_amdgcn_global_load_async_to_lds_b128((av4i*)src, (av4i*)dst, 0, 0);
  __builtin_amdgcn_global_load_async_to_lds_b128((av4i*)(src + 8), (av4i*)(dst + 8), 0, 0);
}
#else
__device__ __forceinline__ void load_tile_sync(unsigned short* __restrict__ lds,
                                               const unsigned short* __restrict__ g,
                                               int t) {
  const int row = t >> 2;
  const int seg = (t & 3) * 16;
  const uint4* src = (const uint4*)(g + (size_t)row * H + seg);
  uint4 a = src[0], b = src[1];
  *(uint4*)&lds[(size_t)row * BKP + seg]     = a;
  *(uint4*)&lds[(size_t)row * BKP + seg + 8] = b;
}
#endif

// A fragment (16x32 bf16): lanes 0-15 carry K 0..7 / 16..23,
// lanes 16-31 carry K 8..15 / 24..31 (ISA 16-bit A layout).
__device__ __forceinline__ v16bf load_a_frag(const unsigned short* __restrict__ lds,
                                             int mblk, int ks, int lane) {
  const int m = mblk * 16 + (lane & 15);
  const int half = lane >> 4;
  const unsigned short* p = lds + (size_t)m * BKP + ks * 32 + half * 8;
  Frag f;
  f.q[0] = *(const float4*)(p);
  f.q[1] = *(const float4*)(p + 16);
  return f.v;
}

// B fragment (32x16 bf16): lanes 0-15 carry K 0..7 / 16..23,
// lanes 16-31 carry K 8..15 / 24..31 (same order as A).
__device__ __forceinline__ v16bf load_b_frag(const unsigned short* __restrict__ lds,
                                             int nblk, int ks, int lane) {
  const int n = nblk * 16 + (lane & 15);
  const int half = lane >> 4;
  const unsigned short* p = lds + (size_t)n * BKP + ks * 32 + half * 8;
  Frag f;
  f.q[0] = *(const float4*)(p);
  f.q[1] = *(const float4*)(p + 16);
  return f.v;
}

__device__ __forceinline__ void mma_step(const unsigned short* As_, const unsigned short* At_,
                                         const unsigned short* Bs_, const unsigned short* Bt_,
                                         int mw, int nw, int lane, v8f acc[2][2]) {
  #pragma unroll
  for (int ks = 0; ks < 2; ++ks) {
    v16bf a[2], b0[2], b1[2];
    a[0]  = load_a_frag(As_, mw, ks, lane);
    a[1]  = load_a_frag(At_, mw, ks, lane);
    b0[0] = load_b_frag(Bs_, nw * 2 + 0, ks, lane);
    b1[0] = load_b_frag(Bs_, nw * 2 + 1, ks, lane);
    b0[1] = load_b_frag(Bt_, nw * 2 + 0, ks, lane);
    b1[1] = load_b_frag(Bt_, nw * 2 + 1, ks, lane);
    #pragma unroll
    for (int mat = 0; mat < 2; ++mat) {
      acc[mat][0] = __builtin_amdgcn_wmma_f32_16x16x32_bf16(
          false, a[mat], false, b0[mat], (short)0, acc[mat][0], false, false);
      acc[mat][1] = __builtin_amdgcn_wmma_f32_16x16x32_bf16(
          false, a[mat], false, b1[mat], (short)0, acc[mat][1], false, false);
    }
  }
}

// PASS==1: per-chunk streaming logsumexp partials.
// PASS==3: recompute logits, accumulate JSD, atomic add to out.
template <int PASS>
__global__ __launch_bounds__(256) void jsd_gemm_kernel(
    const unsigned short* __restrict__ xs, const unsigned short* __restrict__ xt,
    const unsigned short* __restrict__ wsp, const unsigned short* __restrict__ wtp,
    float* __restrict__ lse_part,       // PASS1: [2][NSPLIT][BT]
    const float* __restrict__ Zrow,     // PASS3: [2][BT]
    float* __restrict__ out) {
  __shared__ unsigned short As[2][BM * BKP], At[2][BM * BKP];
  __shared__ unsigned short Bs[2][BN * BKP], Bt[2][BN * BKP];
  __shared__ float tstat_m[2][2][BM], tstat_s[2][2][BM];
  __shared__ float run_m[2][BM], run_s[2][BM];
  __shared__ float zsh[2][BM];
  __shared__ float redbuf[8];

  const int t = threadIdx.x;
  const int lane = t & 31;
  const int wv = t >> 5;
  const int mw = wv & 3;       // 4 m-blocks of 16 rows
  const int nw = wv >> 2;      // 2 n-halves of 32 cols
  const int half = lane >> 4;
  const int row0 = blockIdx.x * BM;
  const int col0 = blockIdx.y * VC;

  if constexpr (PASS == 1) {
    if (t < 2 * BM) {
      const int mat = t >> 6, r = t & 63;
      run_m[mat][r] = -__builtin_huge_valf();
      run_s[mat][r] = 0.f;
    }
  } else {
    if (t < 2 * BM) {
      const int mat = t >> 6, r = t & 63;
      zsh[mat][r] = Zrow[(size_t)mat * BT + row0 + r];
    }
  }
  float lacc = 0.f;
  __syncthreads();

  for (int nt = 0; nt < NTILES; ++nt) {
    const int nbase = col0 + nt * BN;
    const unsigned short* xs0 = xs  + (size_t)row0  * H;
    const unsigned short* xt0 = xt  + (size_t)row0  * H;
    const unsigned short* ws0 = wsp + (size_t)nbase * H;
    const unsigned short* wt0 = wtp + (size_t)nbase * H;

    v8f acc[2][2];
    #pragma unroll
    for (int i = 0; i < 2; ++i)
      #pragma unroll
      for (int j = 0; j < 2; ++j) acc[i][j] = (v8f){};

#if HAVE_ASYNC
    // Prime buffer 0, then ping-pong: prefetch k+1 while computing k.
    load_tile_async(As[0], xs0, t);
    load_tile_async(At[0], xt0, t);
    load_tile_async(Bs[0], ws0, t);
    load_tile_async(Bt[0], wt0, t);
    for (int kt = 0; kt < KITERS; ++kt) {
      const int cur = kt & 1;
      if (kt + 1 < KITERS) {
        const int k1 = (kt + 1) * BK;
        load_tile_async(As[cur ^ 1], xs0 + k1, t);
        load_tile_async(At[cur ^ 1], xt0 + k1, t);
        load_tile_async(Bs[cur ^ 1], ws0 + k1, t);
        load_tile_async(Bt[cur ^ 1], wt0 + k1, t);
        asm volatile("s_wait_asynccnt 0x8" ::: "memory");   // current tile landed
      } else {
        asm volatile("s_wait_asynccnt 0x0" ::: "memory");
      }
      __syncthreads();
      mma_step(As[cur], At[cur], Bs[cur], Bt[cur], mw, nw, lane, acc);
      __syncthreads();
    }
#else
    for (int kt = 0; kt < KITERS; ++kt) {
      const int k0 = kt * BK;
      load_tile_sync(As[0], xs0 + k0, t);
      load_tile_sync(At[0], xt0 + k0, t);
      load_tile_sync(Bs[0], ws0 + k0, t);
      load_tile_sync(Bt[0], wt0 + k0, t);
      __syncthreads();
      mma_step(As[0], At[0], Bs[0], Bt[0], mw, nw, lane, acc);
      __syncthreads();
    }
#endif

    if constexpr (PASS == 1) {
      // Per-row (max, sum exp) over this 64-wide n-tile.
      #pragma unroll
      for (int mat = 0; mat < 2; ++mat) {
        #pragma unroll
        for (int r = 0; r < 8; ++r) {
          float v0 = acc[mat][0][r], v1 = acc[mat][1][r];
          float mv = fmaxf(v0, v1);
          #pragma unroll
          for (int off = 8; off >= 1; off >>= 1)
            mv = fmaxf(mv, __shfl_xor(mv, off, 32));
          float e = __expf(v0 - mv) + __expf(v1 - mv);
          #pragma unroll
          for (int off = 8; off >= 1; off >>= 1)
            e += __shfl_xor(e, off, 32);
          if ((lane & 15) == 0) {
            const int row = mw * 16 + r + half * 8;
            tstat_m[mat][nw][row] = mv;
            tstat_s[mat][nw][row] = e;
          }
        }
      }
      __syncthreads();
      if (t < 2 * BM) {  // merge the two n-halves, then online-merge into run
        const int mat = t >> 6, r = t & 63;
        float m0 = tstat_m[mat][0][r], s0 = tstat_s[mat][0][r];
        float m1 = tstat_m[mat][1][r], s1 = tstat_s[mat][1][r];
        float mm = fmaxf(m0, m1);
        float ss = s0 * __expf(m0 - mm) + s1 * __expf(m1 - mm);
        float rm = run_m[mat][r], rs = run_s[mat][r];
        float nm = fmaxf(rm, mm);
        run_s[mat][r] = rs * __expf(rm - nm) + ss * __expf(mm - nm);
        run_m[mat][r] = nm;
      }
      __syncthreads();
    } else {
      // JSD contribution for every logit element of this tile.
      #pragma unroll
      for (int r = 0; r < 8; ++r) {
        const int row = mw * 16 + r + half * 8;
        const float zq = zsh[0][row];  // student logZ
        const float zp = zsh[1][row];  // teacher logZ
        #pragma unroll
        for (int j = 0; j < 2; ++j) {
          const float lq = acc[0][j][r] - zq;
          const float lp = acc[1][j][r] - zp;
          const float p = __expf(lp), q = __expf(lq);
          const float mm = 0.5f * (p + q);
          const float lm = __logf(mm);
          lacc += 0.5f * (p * (lp - lm) + q * (lq - lm));
        }
      }
    }
  }

  if constexpr (PASS == 1) {
    if (t < 2 * BM) {
      const int mat = t >> 6, r = t & 63;
      const float lse = run_m[mat][r] + __logf(run_s[mat][r]);
      lse_part[((size_t)mat * NSPLIT + blockIdx.y) * BT + row0 + r] = lse;
    }
  } else {
    #pragma unroll
    for (int off = 16; off >= 1; off >>= 1) lacc += __shfl_xor(lacc, off, 32);
    if (lane == 0) redbuf[wv] = lacc;
    __syncthreads();
    if (t == 0) {
      float s = 0.f;
      #pragma unroll
      for (int i = 0; i < 8; ++i) s += redbuf[i];
      atomicAdd(out, s * (1.0f / BT));
    }
  }
}

__global__ void lse_reduce_kernel(const float* __restrict__ lse_part,
                                  float* __restrict__ Zrow) {
  const int idx = blockIdx.x * blockDim.x + threadIdx.x;
  if (idx >= 2 * BT) return;
  const int mat = idx >> 12;          // BT == 4096
  const int row = idx & (BT - 1);
  float M = -__builtin_huge_valf();
  for (int c = 0; c < NSPLIT; ++c)
    M = fmaxf(M, lse_part[((size_t)mat * NSPLIT + c) * BT + row]);
  float s = 0.f;
  for (int c = 0; c < NSPLIT; ++c)
    s += __expf(lse_part[((size_t)mat * NSPLIT + c) * BT + row] - M);
  Zrow[idx] = M + __logf(s);
}

__global__ void zero_out_kernel(float* out) { out[0] = 0.f; }

extern "C" void kernel_launch(void* const* d_in, const int* in_sizes, int n_in,
                              void* d_out, int out_size, void* d_ws, size_t ws_size,
                              hipStream_t stream) {
  (void)in_sizes; (void)n_in; (void)out_size; (void)ws_size;
  const float* xs_f  = (const float*)d_in[0];  // student_input [BT,H]
  const float* xt_f  = (const float*)d_in[1];  // teacher_input [BT,H]
  const float* ws_f  = (const float*)d_in[2];  // w_student [V,H]
  const float* wt_f  = (const float*)d_in[3];  // w_teacher [V,H]
  float* out = (float*)d_out;

  // Workspace layout: bf16 copies of all tensors, then reduction scratch.
  unsigned short* wsb = (unsigned short*)d_ws;           // V*H
  unsigned short* wtb = wsb + (size_t)V * H;             // V*H
  unsigned short* xsb = wtb + (size_t)V * H;             // BT*H
  unsigned short* xtb = xsb + (size_t)BT * H;            // BT*H
  float* lse_part = (float*)(xtb + (size_t)BT * H);      // 2*NSPLIT*BT
  float* Zrow     = lse_part + (size_t)2 * NSPLIT * BT;  // 2*BT

  // One-time fp32 -> bf16 conversion (removes pack ALU + halves GEMM traffic).
  cvt_bf16_kernel<<<(V * H) / (256 * 16), 256, 0, stream>>>(ws_f, wsb);
  cvt_bf16_kernel<<<(V * H) / (256 * 16), 256, 0, stream>>>(wt_f, wtb);
  cvt_bf16_kernel<<<(BT * H) / (256 * 16), 256, 0, stream>>>(xs_f, xsb);
  cvt_bf16_kernel<<<(BT * H) / (256 * 16), 256, 0, stream>>>(xt_f, xtb);

  zero_out_kernel<<<1, 1, 0, stream>>>(out);

  dim3 grid(NBLK, NSPLIT);
  jsd_gemm_kernel<1><<<grid, 256, 0, stream>>>(xsb, xtb, wsb, wtb,
                                               lse_part, nullptr, nullptr);
  lse_reduce_kernel<<<(2 * BT + 255) / 256, 256, 0, stream>>>(lse_part, Zrow);
  jsd_gemm_kernel<3><<<grid, 256, 0, stream>>>(xsb, xtb, wsb, wtb,
                                               nullptr, Zrow, out);
}
